// PointEncoder_27650999452600
// MI455X (gfx1250) — matrix operand order repaired and checked by reading, not compiled, not run on hardware
//
#include <hip/hip_runtime.h>

typedef __attribute__((ext_vector_type(16))) _Float16 v16h;
typedef __attribute__((ext_vector_type(8)))  _Float16 v8h;
typedef __attribute__((ext_vector_type(8)))  float    v8f;

#define NPTS 16384
#define BATCH 4
#define RTOT (BATCH * NPTS)

enum { GF_RELU = 1, GF_PRE = 2, GF_POST = 4 };

// ---------------------------------------------------------------------------
// WMMA GEMM:  Y[r, o] = act( sum_k X[r,k] * W[o,k] + bias[o] ) (* dw)
// X: (BATCH*in_N, in_ld) f16 row-major (reads cols [0, CinP))
// W: (Cout, CinP) f16 row-major (zero-padded along K)
// Row slicing: output row (b, i) reads input row b*in_N + i, i < out_N <= in_N.
// One wave = one 16x16 tile, K swept 32 at a time via v_wmma_f32_16x16x32_f16.
// ---------------------------------------------------------------------------
__global__ __launch_bounds__(256) void k_gemm(
    const _Float16* __restrict__ X, int in_ld,
    const _Float16* __restrict__ W, int CinP,
    const float* __restrict__ bias,
    const float* __restrict__ dw,
    _Float16* __restrict__ Y, int out_ld,
    int in_N, int out_N, int Cout, int flags)
{
  const int lane = threadIdx.x & 31;
  const int wv   = threadIdx.x >> 5;
  const int colTiles = Cout >> 4;
  const int rowTiles = (BATCH * out_N) >> 4;
  int tile = blockIdx.x * 8 + wv;
  if (tile >= rowTiles * colTiles) return;
  int rt = tile / colTiles;
  int ct = tile - rt * colTiles;
  int row0 = rt << 4, c0 = ct << 4;
  int b  = row0 / out_N;
  int i0 = row0 - b * out_N;
  long inrow = (long)b * in_N + i0;
  const int kh = lane >> 4;      // half-of-wave selector
  const int nn = lane & 15;      // A-row m (for SRC0) and B/D column n

  // A operand: lane holds row m=nn; half j -> k = kh*16 + j  (contiguous in X row)
  const _Float16* ap = X + (inrow + nn) * (long)in_ld + (kh << 4);
  // B operand: lane holds col n=nn; half j -> k = kh*16 + j  (contiguous in W row)
  const _Float16* bp = W + (long)(c0 + nn) * CinP + (kh << 4);

  _Float16 psc = (_Float16)1.0f;
  if (flags & GF_PRE) psc = (_Float16)dw[b * NPTS + i0 + nn];

  v8f acc = {};
  for (int k0 = 0; k0 < CinP; k0 += 32) {
    union { v16h v; v8h h2[2]; _Float16 e[16]; } A;
    A.h2[0] = *(const v8h*)(ap + k0);
    A.h2[1] = *(const v8h*)(ap + k0 + 8);
    v16h Bm = *(const v16h*)(bp + k0);
    if (flags & GF_PRE) {
      #pragma unroll
      for (int j = 0; j < 16; ++j) A.e[j] = A.e[j] * psc;
    }
    acc = __builtin_amdgcn_wmma_f32_16x16x32_f16(
        false, A.v, false, Bm, (short)0, acc, false, false);
  }

  float bs = bias[c0 + nn];
  #pragma unroll
  for (int r = 0; r < 8; ++r) {
    int mm = (kh << 3) + r;                 // D row for VGPR r
    float v = acc[r] + bs;
    if (flags & GF_RELU) v = fmaxf(v, 0.0f);
    if (flags & GF_POST) v *= dw[b * NPTS + i0 + mm];
    Y[((long)row0 + mm) * out_ld + c0 + nn] = (_Float16)v;
  }
}

// ------------------------- weight convert (f32 -> f16, K-padded) ------------
__global__ __launch_bounds__(256) void k_cvtw(const float* __restrict__ src,
                                              _Float16* __restrict__ dst,
                                              int Cout, int Cin, int CinP)
{
  int t = blockIdx.x * 256 + threadIdx.x;
  if (t >= Cout * CinP) return;
  int o = t / CinP, k = t - o * CinP;
  dst[t] = (k < Cin) ? (_Float16)src[o * Cin + k] : (_Float16)0.0f;
}

// ------------------------- f16 column-range copy (concat builder) -----------
__global__ __launch_bounds__(256) void k_copy16(const _Float16* __restrict__ src,
                                                int sld, int soff,
                                                _Float16* __restrict__ dst,
                                                int dld, int doff,
                                                int rows, int cols)
{
  int t = blockIdx.x * 256 + threadIdx.x;
  if (t >= rows * cols) return;
  int r = t / cols, c = t - r * cols;
  dst[(long)r * dld + doff + c] = src[(long)r * sld + soff + c];
}

__global__ __launch_bounds__(256) void k_cast32to16(const float* __restrict__ src,
                                                    int sld, int soff,
                                                    _Float16* __restrict__ dst,
                                                    int dld, int doff,
                                                    int rows, int cols)
{
  int t = blockIdx.x * 256 + threadIdx.x;
  if (t >= rows * cols) return;
  int r = t / cols, c = t - r * cols;
  dst[(long)r * dld + doff + c] = (_Float16)src[(long)r * sld + soff + c];
}

__global__ __launch_bounds__(256) void k_fill16(_Float16* __restrict__ dst,
                                                int dld, int doff,
                                                int rows, int cols)
{
  int t = blockIdx.x * 256 + threadIdx.x;
  if (t >= rows * cols) return;
  int r = t / cols, c = t - r * cols;
  dst[(long)r * dld + doff + c] = (_Float16)0.0f;
}

// ------------------------- dw = normalized sigmoid distance weight ----------
__global__ __launch_bounds__(256) void k_dw_raw(const float* __restrict__ pos,
                                                const float* __restrict__ alpha,
                                                const float* __restrict__ beta,
                                                float* __restrict__ dwr)
{
  int t = blockIdx.x * 256 + threadIdx.x;
  if (t >= RTOT) return;
  int b = t >> 14, i = t & (NPTS - 1);
  const float* p = pos + (long)b * 3 * NPTS;
  float x = p[i], y = p[NPTS + i], z = p[2 * NPTS + i];
  float d = sqrtf(x * x + y * y + z * z);
  float u = -alpha[0] * d + beta[0];
  dwr[t] = 1.0f / (1.0f + expf(-u));
}

__global__ __launch_bounds__(256) void k_dw_sum(const float* __restrict__ dwr,
                                                float* __restrict__ ss)
{
  __shared__ float red[256];
  int b = blockIdx.x;
  float s = 0.0f;
  for (int j = threadIdx.x; j < NPTS; j += 256) s += dwr[b * NPTS + j];
  red[threadIdx.x] = s;
  __syncthreads();
  for (int st = 128; st > 0; st >>= 1) {
    if ((int)threadIdx.x < st) red[threadIdx.x] += red[threadIdx.x + st];
    __syncthreads();
  }
  if (threadIdx.x == 0) ss[b] = red[0];
}

__global__ __launch_bounds__(256) void k_dw_norm(float* __restrict__ dwb,
                                                 const float* __restrict__ ss)
{
  int t = blockIdx.x * 256 + threadIdx.x;
  if (t >= RTOT) return;
  float s = ss[t >> 14];
  float s2 = s + ((s == 0.0f) ? 1.0f : 0.0f) + 1e-06f;
  dwb[t] = dwb[t] / s2 * (float)NPTS;
}

// ------------------------- encoder layer 1: 6 -> 24 + max over K ------------
__global__ __launch_bounds__(256) void k_enc1(const float* __restrict__ pos,
                                              const int* __restrict__ knn,
                                              const float* __restrict__ W,
                                              const float* __restrict__ bias,
                                              float* __restrict__ h1)
{
  __shared__ float sW[144];
  __shared__ float sB[24];
  for (int t = threadIdx.x; t < 144; t += 256) sW[t] = W[t];
  if (threadIdx.x < 24) sB[threadIdx.x] = bias[threadIdx.x];
  __syncthreads();
  int t = blockIdx.x * 256 + threadIdx.x;
  if (t >= RTOT) return;
  int b = t >> 14, i = t & (NPTS - 1);
  const float* pb = pos + (long)b * 3 * NPTS;
  float cx = pb[i], cy = pb[NPTS + i], cz = pb[2 * NPTS + i];
  float acc[24];
  #pragma unroll
  for (int o = 0; o < 24; ++o) acc[o] = 0.0f;   // relu >= 0, K>=1
  const int* kp = knn + (long)t * 16;
  for (int k = 0; k < 16; ++k) {
    int idx = kp[k];
    float nx = pb[idx] - cx, ny = pb[NPTS + idx] - cy, nz = pb[2 * NPTS + idx] - cz;
    #pragma unroll
    for (int o = 0; o < 24; ++o) {
      const float* w = sW + o * 6;
      float v = sB[o] + cx * w[0] + cy * w[1] + cz * w[2]
                      + nx * w[3] + ny * w[4] + nz * w[5];
      acc[o] = fmaxf(acc[o], v);
    }
  }
  float* hp = h1 + (long)t * 24;
  #pragma unroll
  for (int o = 0; o < 24; ++o) hp[o] = acc[o];
}

// --------- encoder layer 2: gather 24ch, edge 48 -> 24 + max over K ---------
// v(o,k) = bias[o] + sum_c ctr[c]*(W1-W2)[o,c] + sum_c nb[c]*W2[o,c]
__global__ __launch_bounds__(256) void k_enc2(const float* __restrict__ h1,
                                              const int* __restrict__ knn,
                                              const float* __restrict__ W,
                                              const float* __restrict__ bias,
                                              _Float16* __restrict__ code)
{
  __shared__ float sW2[576];   // W[o][24+c]
  __shared__ float sWd[576];   // W[o][c] - W[o][24+c]
  __shared__ float sB[24];
  for (int t = threadIdx.x; t < 576; t += 256) {
    int o = t / 24, c = t - o * 24;
    float w2 = W[o * 48 + 24 + c];
    sW2[t] = w2;
    sWd[t] = W[o * 48 + c] - w2;
  }
  if (threadIdx.x < 24) sB[threadIdx.x] = bias[threadIdx.x];
  __syncthreads();
  int t = blockIdx.x * 256 + threadIdx.x;
  if (t >= RTOT) return;
  int b = t >> 14;
  float ctr[24];
  const float* hp = h1 + (long)t * 24;
  #pragma unroll
  for (int c = 0; c < 24; ++c) ctr[c] = hp[c];
  float base[24], acc[24];
  #pragma unroll
  for (int o = 0; o < 24; ++o) {
    float s = sB[o];
    for (int c = 0; c < 24; ++c) s += ctr[c] * sWd[o * 24 + c];
    base[o] = s;
    acc[o] = 0.0f;
  }
  const int* kp = knn + (long)t * 16;
  const float* hb = h1 + (long)b * NPTS * 24;
  for (int k = 0; k < 16; ++k) {
    int idx = kp[k];
    const float* np = hb + (long)idx * 24;
    float nb[24];
    #pragma unroll
    for (int c = 0; c < 24; ++c) nb[c] = np[c];
    #pragma unroll
    for (int o = 0; o < 24; ++o) {
      float v = base[o];
      for (int c = 0; c < 24; ++c) v += nb[c] * sW2[o * 24 + c];
      acc[o] = fmaxf(acc[o], v);
    }
  }
  _Float16* cp = code + (long)t * 48 + 24;
  #pragma unroll
  for (int o = 0; o < 24; ++o) cp[o] = (_Float16)acc[o];
}

// ------------------------- mlp1 layer 1 (Cin=3, scalar) ---------------------
__global__ __launch_bounds__(256) void k_mlp1a(const float* __restrict__ pos,
                                               const float* __restrict__ W,
                                               const float* __restrict__ bias,
                                               _Float16* __restrict__ p1)
{
  __shared__ float sW[192];
  __shared__ float sB[64];
  for (int t = threadIdx.x; t < 192; t += 256) sW[t] = W[t];
  if (threadIdx.x < 64) sB[threadIdx.x] = bias[threadIdx.x];
  __syncthreads();
  int t = blockIdx.x * 256 + threadIdx.x;
  if (t >= RTOT) return;
  int b = t >> 14, i = t & (NPTS - 1);
  const float* pb = pos + (long)b * 3 * NPTS;
  float x = pb[i], y = pb[NPTS + i], z = pb[2 * NPTS + i];
  _Float16* pp = p1 + (long)t * 64;
  #pragma unroll
  for (int o = 0; o < 64; ++o) {
    float v = sB[o] + x * sW[o * 3] + y * sW[o * 3 + 1] + z * sW[o * 3 + 2];
    pp[o] = (_Float16)fmaxf(v, 0.0f);
  }
}

// ------------------------- outputs: transpose (B,N',C) -> (B,C,N') ----------
__global__ __launch_bounds__(256) void k_out_t(const _Float16* __restrict__ src,
                                               int src_ld, int srcN,
                                               float* __restrict__ dst,
                                               int C, int No)
{
  int t = blockIdx.x * 256 + threadIdx.x;
  if (t >= BATCH * C * No) return;
  int b = t / (C * No);
  int rem = t - b * C * No;
  int c = rem / No, i = rem - c * No;
  dst[t] = (float)src[((long)b * srcN + i) * src_ld + c];
}

__global__ __launch_bounds__(256) void k_out_dw(const float* __restrict__ dw,
                                                float* __restrict__ dst, int No)
{
  int t = blockIdx.x * 256 + threadIdx.x;
  if (t >= BATCH * No) return;
  int b = t / No, i = t - b * No;
  dst[t] = dw[b * NPTS + i];
}

// ---------------------------------------------------------------------------
static inline void gemm(const _Float16* X, int in_ld, const _Float16* W, int CinP,
                        const float* bias, const float* dw,
                        _Float16* Y, int out_ld,
                        int in_N, int out_N, int Cout, int flags, hipStream_t s)
{
  int tiles = ((BATCH * out_N) >> 4) * (Cout >> 4);
  int blocks = (tiles + 7) / 8;
  k_gemm<<<blocks, 256, 0, s>>>(X, in_ld, W, CinP, bias, dw, Y, out_ld,
                                in_N, out_N, Cout, flags);
}

static inline int gsz(long n) { return (int)((n + 255) / 256); }

extern "C" void kernel_launch(void* const* d_in, const int* in_sizes, int n_in,
                              void* d_out, int out_size, void* d_ws, size_t ws_size,
                              hipStream_t stream)
{
  (void)in_sizes; (void)n_in; (void)out_size; (void)ws_size;
  const float* pos   = (const float*)d_in[0];
  const int*   knn   = (const int*)d_in[1];
  const float* encW0 = (const float*)d_in[2];
  const float* encW1 = (const float*)d_in[3];
  const float* encB0 = (const float*)d_in[4];
  const float* encB1 = (const float*)d_in[5];
  const float* m1W0  = (const float*)d_in[6];
  const float* m1W1  = (const float*)d_in[7];
  const float* m1B0  = (const float*)d_in[8];
  const float* m1B1  = (const float*)d_in[9];
  const float* linW[7]; const float* linB[7];
  for (int i = 0; i < 7; ++i) { linW[i] = (const float*)d_in[10 + i]; linB[i] = (const float*)d_in[17 + i]; }
  const float* c1W = (const float*)d_in[24];
  const float* c1B = (const float*)d_in[25];
  const float* pbW[4]; const float* pbB[4];
  for (int i = 0; i < 4; ++i) { pbW[i] = (const float*)d_in[26 + i]; pbB[i] = (const float*)d_in[30 + i]; }
  const float* c2W = (const float*)d_in[34];
  const float* c2B = (const float*)d_in[35];
  const float* c3W = (const float*)d_in[36];
  const float* c3B = (const float*)d_in[37];
  const float* alpha = (const float*)d_in[38];
  const float* beta  = (const float*)d_in[39];
  float* out = (float*)d_out;

  char* ws = (char*)d_ws;
  size_t off = 0;
  auto A = [&](size_t n) -> char* {
    off = (off + 255) & ~(size_t)255;
    char* p = ws + off;
    off += n;
    return p;
  };
  const long R = RTOT;

  // f16 weights (K padded to multiple of 32)
  _Float16* wg_m1b = (_Float16*)A(64 * 64 * 2);
  _Float16* wg_lin[7];
  const int linCout[7] = {128, 128, 128, 16, 128, 128, 64};
  for (int i = 0; i < 7; ++i) wg_lin[i] = (_Float16*)A((size_t)linCout[i] * 128 * 2);
  _Float16* wg_c1 = (_Float16*)A(128 * 128 * 2);
  _Float16* wg_pb[4];
  wg_pb[0] = (_Float16*)A(256 * 128 * 2);
  for (int i = 1; i < 4; ++i) wg_pb[i] = (_Float16*)A(256 * 256 * 2);
  _Float16* wg_c2 = (_Float16*)A(256 * 256 * 2);
  _Float16* wg_c3 = (_Float16*)A(128 * 256 * 2);

  // activations
  float*    h1   = (float*)A(R * 24 * 4);
  _Float16* code = (_Float16*)A(R * 48 * 2);
  _Float16* p1   = (_Float16*)A(R * 64 * 2);
  _Float16* p    = (_Float16*)A(R * 64 * 2);
  _Float16* z0   = (_Float16*)A(R * 128 * 2);   // also reused as yin
  _Float16* zA   = (_Float16*)A(R * 128 * 2);
  _Float16* zB   = (_Float16*)A(R * 128 * 2);
  _Float16* y0   = (_Float16*)A(R * 64 * 2);
  _Float16* y    = (_Float16*)A(R * 128 * 2);
  float*    dwb  = (float*)A(R * 4);
  float*    ssum = (float*)A(BATCH * 4);
  // aliases for the shrinking pb chain (previous buffers dead by then)
  _Float16* t0 = zA;   // (B*8192, 256)  = 16.8MB <= 16.8MB
  _Float16* t1 = zB;   // (B*8192, 256)
  _Float16* t2 = p1;   // (B*4096, 256)  = 8.4MB <= 8.4MB
  _Float16* t3 = p;    // (B*4096, 256)
  _Float16* t4 = y0;   // (B*4096, 256)
  _Float16* t5 = (_Float16*)h1; // (B*4096, 128) = 4.2MB <= 6.3MB

  // ---- weight conversion -------------------------------------------------
  k_cvtw<<<gsz(64 * 64), 256, 0, stream>>>(m1W1, wg_m1b, 64, 64, 64);
  const int linCin[7] = {112, 128, 128, 128, 128, 128, 128};
  for (int i = 0; i < 7; ++i)
    k_cvtw<<<gsz((long)linCout[i] * 128), 256, 0, stream>>>(linW[i], wg_lin[i], linCout[i], linCin[i], 128);
  k_cvtw<<<gsz(128 * 128), 256, 0, stream>>>(c1W, wg_c1, 128, 112, 128);
  k_cvtw<<<gsz(256 * 128), 256, 0, stream>>>(pbW[0], wg_pb[0], 256, 128, 128);
  for (int i = 1; i < 4; ++i)
    k_cvtw<<<gsz(256 * 256), 256, 0, stream>>>(pbW[i], wg_pb[i], 256, 256, 256);
  k_cvtw<<<gsz(256 * 256), 256, 0, stream>>>(c2W, wg_c2, 256, 256, 256);
  k_cvtw<<<gsz(128 * 256), 256, 0, stream>>>(c3W, wg_c3, 128, 256, 256);

  // ---- dw ----------------------------------------------------------------
  k_dw_raw<<<gsz(R), 256, 0, stream>>>(pos, alpha, beta, dwb);
  k_dw_sum<<<BATCH, 256, 0, stream>>>(dwb, ssum);
  k_dw_norm<<<gsz(R), 256, 0, stream>>>(dwb, ssum);

  // ---- encoder -----------------------------------------------------------
  k_enc1<<<gsz(R), 256, 0, stream>>>(pos, knn, encW0, encB0, h1);
  k_cast32to16<<<gsz(R * 24), 256, 0, stream>>>(h1, 24, 0, code, 48, 0, (int)R, 24);
  k_enc2<<<gsz(R), 256, 0, stream>>>(h1, knn, encW1, encB1, code);

  // ---- mlp1 --------------------------------------------------------------
  k_mlp1a<<<gsz(R), 256, 0, stream>>>(pos, m1W0, m1B0, p1);
  gemm(p1, 64, wg_m1b, 64, m1B1, nullptr, p, 64, NPTS, NPTS, 64, 0, stream); // no relu

  // ---- z0 = [p(64) | code(48) | zero(16)] --------------------------------
  k_fill16<<<gsz(R * 16), 256, 0, stream>>>(z0, 128, 112, (int)R, 16);
  k_copy16<<<gsz(R * 64), 256, 0, stream>>>(p, 64, 0, z0, 128, 0, (int)R, 64);
  k_copy16<<<gsz(R * 48), 256, 0, stream>>>(code, 48, 0, z0, 128, 64, (int)R, 48);

  // ---- lin chain ---------------------------------------------------------
  gemm(z0, 128, wg_lin[0], 128, linB[0], nullptr, zA, 128, NPTS, NPTS, 128, GF_RELU, stream);
  gemm(zA, 128, wg_lin[1], 128, linB[1], nullptr, zB, 128, NPTS, NPTS, 128, GF_RELU, stream);
  gemm(zB, 128, wg_lin[2], 128, linB[2], nullptr, zA, 128, NPTS, NPTS, 128, GF_RELU, stream);
  gemm(zA, 128, wg_lin[3], 128, linB[3], nullptr, zB, 128, NPTS, NPTS, 16, GF_RELU, stream);
  // zB = [z16(16) | p(64) | code(48)]
  k_copy16<<<gsz(R * 64), 256, 0, stream>>>(p, 64, 0, zB, 128, 16, (int)R, 64);
  k_copy16<<<gsz(R * 48), 256, 0, stream>>>(code, 48, 0, zB, 128, 80, (int)R, 48);
  gemm(zB, 128, wg_lin[4], 128, linB[4], nullptr, zA, 128, NPTS, NPTS, 128, GF_RELU, stream);
  gemm(zA, 128, wg_lin[5], 128, linB[5], nullptr, zB, 128, NPTS, NPTS, 128, GF_RELU, stream);
  gemm(zB, 128, wg_lin[6], 128, linB[6], nullptr, y0, 64, NPTS, NPTS, 64, 0, stream); // no relu

  // ---- y = relu(c1([code | y0])) ; reuse z0 as yin (pad cols still zero) --
  k_copy16<<<gsz(R * 48), 256, 0, stream>>>(code, 48, 0, z0, 128, 0, (int)R, 48);
  k_copy16<<<gsz(R * 64), 256, 0, stream>>>(y0, 64, 0, z0, 128, 48, (int)R, 64);
  gemm(z0, 128, wg_c1, 128, c1B, nullptr, y, 128, NPTS, NPTS, 128, GF_RELU, stream);

  // ---- pb chain (row-sliced, dw-scaled) ----------------------------------
  gemm(y,  128, wg_pb[0], 128, pbB[0], dwb, t0, 256, NPTS,     NPTS / 2, 256, GF_RELU | GF_POST, stream);
  gemm(t0, 256, wg_pb[1], 256, pbB[1], dwb, t1, 256, NPTS / 2, NPTS / 2, 256, GF_RELU | GF_POST, stream);
  gemm(t1, 256, wg_pb[2], 256, pbB[2], dwb, t2, 256, NPTS / 2, NPTS / 4, 256, GF_RELU | GF_POST, stream);
  gemm(t2, 256, wg_pb[3], 256, pbB[3], dwb, t3, 256, NPTS / 4, NPTS / 4, 256, GF_RELU | GF_PRE,  stream);
  gemm(t3, 256, wg_c2,    256, c2B,  nullptr, t4, 256, NPTS / 4, NPTS / 4, 256, GF_RELU, stream);
  gemm(t4, 256, wg_c3,    256, c3B,  nullptr, t5, 128, NPTS / 4, NPTS / 4, 128, GF_RELU, stream);

  // ---- outputs -----------------------------------------------------------
  const int No = NPTS / 4;
  k_out_t<<<gsz((long)BATCH * 128 * No), 256, 0, stream>>>(t5, 128, No,   out,                 128, No);
  k_out_t<<<gsz((long)BATCH * 128 * No), 256, 0, stream>>>(y,  128, NPTS, out + 2097152,       128, No);
  k_out_dw<<<gsz((long)BATCH * No), 256, 0, stream>>>(dwb, out + 4194304, No);
}
